// AudioModelV5_30709016166498
// MI455X (gfx1250) — compile-verified, matrix operand order checked by reading, not run on hardware
//
#include <hip/hip_runtime.h>
#include <hip/hip_bf16.h>
#include <math.h>

typedef __attribute__((ext_vector_type(16))) __bf16 v16bf;
typedef __attribute__((ext_vector_type(8)))  float  v8f;

#define D_MODEL 1024
#define D_INNER 2048
#define N_STATE 64
#define HEADS   32
#define PDIM    64
#define CONV_DIM (D_INNER + 2*N_STATE)         /* 2176 */
#define D_CONV  4
#define N_LAYERS 5
#define BATCH   4
#define SEQ     1024
#define TOKENS  (BATCH*SEQ)                    /* 4096 */
#define NPROJ   (2*D_INNER + 2*N_STATE + HEADS)/* 4256 */

// register-blocking factors for the WMMA GEMM (per wave: 64x32 output block)
#define MT 4
#define NT 2

__device__ __forceinline__ unsigned short f32_to_bf16_rne(float f) {
    unsigned int u = __float_as_uint(f);
    unsigned int r = 0x7FFFu + ((u >> 16) & 1u);
    u += r;
    return (unsigned short)(u >> 16);
}

// ---------------------------------------------------------------- convert
__global__ void cvt_bf16_kernel(const float* __restrict__ src,
                                unsigned short* __restrict__ dst, int n) {
    int i = blockIdx.x * blockDim.x + threadIdx.x;
    if (i < n) dst[i] = f32_to_bf16_rne(src[i]);
}

// ---------------------------------------------------------------- WMMA GEMM
// C[M,N] = A[M,K](bf16,row-major) * Bw[N,K](bf16,row-major)^T  (fp32 accum)
// Each wave computes an MT x NT grid of 16x16 tiles (register blocking):
// per 32-wide K step: MT+NT fragment loads feed MT*NT WMMAs.
// epilogue 0: plain f32 store.  epilogue 1: SELU, store f32 + bf16 mirror.
__global__ __launch_bounds__(256)
void wmma_gemm_bf16_kernel(const unsigned short* __restrict__ A,
                           const unsigned short* __restrict__ Bw,
                           float* __restrict__ C,
                           unsigned short* __restrict__ Cbf,
                           int M, int N, int K, int epilogue) {
    const int lane = threadIdx.x & 31;
    const int wave = threadIdx.x >> 5;
    const int ntn  = N / (16 * NT);          // N super-tiles
    const int wg   = blockIdx.x * (blockDim.x >> 5) + wave;
    const int total = (M / (16 * MT)) * ntn;
    if (wg >= total) return;                 // wave-uniform: EXEC all-1 for WMMA
    const int tm = (wg / ntn) * MT;          // base 16-row tile index
    const int tn = (wg % ntn) * NT;          // base 16-col tile index

    // A fragment: lane<16 -> row=lane, K in {0..7}U{16..23}; lanes 16..31 the rest.
    const int row   = lane & 15;
    const int khalf = (lane >> 4) << 3;      // 0 or 8

    const unsigned short* Ap[MT];
    const unsigned short* Bp[NT];
    #pragma unroll
    for (int i = 0; i < MT; ++i)
        Ap[i] = A  + (size_t)((tm + i) * 16 + row) * K + khalf;
    #pragma unroll
    for (int j = 0; j < NT; ++j)
        Bp[j] = Bw + (size_t)((tn + j) * 16 + row) * K + khalf;

    v8f acc[MT][NT];
    #pragma unroll
    for (int i = 0; i < MT; ++i)
        #pragma unroll
        for (int j = 0; j < NT; ++j)
            acc[i][j] = (v8f){};

    union frag { v16bf v; uint4 u[2]; };

    for (int k0 = 0; k0 < K; k0 += 32) {
        frag af[MT], bf[NT];
        #pragma unroll
        for (int i = 0; i < MT; ++i) {
            af[i].u[0] = *(const uint4*)(Ap[i] + k0);
            af[i].u[1] = *(const uint4*)(Ap[i] + k0 + 16);
        }
        #pragma unroll
        for (int j = 0; j < NT; ++j) {
            bf[j].u[0] = *(const uint4*)(Bp[j] + k0);
            bf[j].u[1] = *(const uint4*)(Bp[j] + k0 + 16);
        }
        if (k0 + 256 < K) {                  // warm L2/L0 ahead of the pipeline
            __builtin_prefetch(Ap[0] + k0 + 256, 0, 3);
            __builtin_prefetch(Bp[0] + k0 + 256, 0, 3);
        }
        #pragma unroll
        for (int i = 0; i < MT; ++i)
            #pragma unroll
            for (int j = 0; j < NT; ++j)
                acc[i][j] = __builtin_amdgcn_wmma_f32_16x16x32_bf16(
                    /*neg_a=*/false, af[i].v, /*neg_b=*/false, bf[j].v,
                    /*c_mod=*/(short)0, acc[i][j],
                    /*reuse_a=*/false, /*reuse_b=*/false);
    }

    // D layout: VGPR r, lanes 0-15 -> M=r, lanes 16-31 -> M=r+8; N = lane&15.
    const int mb  = (lane >> 4) << 3;
    const int nnl = lane & 15;
    #pragma unroll
    for (int i = 0; i < MT; ++i) {
        #pragma unroll
        for (int j = 0; j < NT; ++j) {
            #pragma unroll
            for (int r = 0; r < 8; ++r) {
                float v = acc[i][j][r];
                if (epilogue == 1) {
                    v = 1.0507009873554805f *
                        (v > 0.f ? v : 1.6732632423543772f * (__expf(v) - 1.f));
                }
                size_t idx = (size_t)((tm + i) * 16 + mb + r) * N +
                             (tn + j) * 16 + nnl;
                C[idx] = v;
                if (epilogue == 1) Cbf[idx] = f32_to_bf16_rne(v);
            }
        }
    }
}

// ---------------------------------------------------------------- conv + SiLU
__global__ void conv_silu_kernel(const float* __restrict__ zx,   // (TOKENS,NPROJ)
                                 const float* __restrict__ cw,   // (CONV_DIM,4)
                                 const float* __restrict__ cb,   // (CONV_DIM)
                                 float* __restrict__ xbc) {      // (TOKENS,CONV_DIM)
    int idx = blockIdx.x * blockDim.x + threadIdx.x;
    if (idx >= TOKENS * CONV_DIM) return;
    int c   = idx % CONV_DIM;
    int tok = idx / CONV_DIM;
    int l = tok % SEQ, b = tok / SEQ;
    float acc = cb[c];
    #pragma unroll
    for (int j = 0; j < D_CONV; ++j) {
        int t = l - (D_CONV - 1) + j;
        if (t >= 0)
            acc += cw[c * D_CONV + j] *
                   zx[((size_t)(b * SEQ + t)) * NPROJ + D_INNER + c];
    }
    xbc[(size_t)tok * CONV_DIM + c] = acc / (1.f + __expf(-acc));  // silu
}

// ---------------------------------------------------------------- softplus(dt)
__global__ void dt_kernel(const float* __restrict__ zx,
                          const float* __restrict__ dtb,
                          float* __restrict__ dt) {
    int idx = blockIdx.x * blockDim.x + threadIdx.x;
    if (idx >= TOKENS * HEADS) return;
    int h = idx % HEADS;
    int tok = idx / HEADS;
    float v = zx[(size_t)tok * NPROJ + (NPROJ - HEADS) + h] + dtb[h];
    dt[idx] = (v > 20.f) ? v : log1pf(__expf(v));
}

// ---------------------------------------------------------------- selective scan
// One block per (batch, head). 256 threads: tid = p*4 + j; thread owns 16 of the
// 64 state columns for its p. State in registers; B/C/x staged via LDS per step.
__global__ __launch_bounds__(256)
void scan_kernel(const float* __restrict__ xbc,   // (TOKENS,CONV_DIM)
                 const float* __restrict__ dt,    // (TOKENS,HEADS)
                 const float* __restrict__ alog,  // (HEADS)
                 const float* __restrict__ Dp,    // (HEADS)
                 float* __restrict__ y) {         // (TOKENS,D_INNER)
    const int b = blockIdx.x / HEADS;
    const int h = blockIdx.x % HEADS;
    const int tid = threadIdx.x;
    const int p = tid >> 2;
    const int j = tid & 3;
    const int nbase = j * 16;
    const float Ah = -__expf(alog[h]);
    const float Dh = Dp[h];

    float hreg[16];
    #pragma unroll
    for (int i = 0; i < 16; ++i) hreg[i] = 0.f;

    __shared__ float sB[N_STATE], sC[N_STATE], sX[PDIM], sdt[1];

    for (int t = 0; t < SEQ; ++t) {
        size_t tok = (size_t)b * SEQ + t;
        const float* base = xbc + tok * CONV_DIM;
        if (tid < 64)       sB[tid]       = base[D_INNER + tid];
        else if (tid < 128) sC[tid - 64]  = base[D_INNER + N_STATE + (tid - 64)];
        else if (tid < 192) sX[tid - 128] = base[h * PDIM + (tid - 128)];
        else if (tid == 192) sdt[0] = dt[tok * HEADS + h];
        __syncthreads();

        float dtv = sdt[0];
        float dA  = __expf(dtv * Ah);
        float xv  = sX[p];
        float dtx = dtv * xv;
        float partial = 0.f;
        #pragma unroll
        for (int i = 0; i < 16; ++i) {
            float hn = hreg[i] * dA + dtx * sB[nbase + i];
            hreg[i] = hn;
            partial += sC[nbase + i] * hn;
        }
        partial += __shfl_xor(partial, 1, 32);
        partial += __shfl_xor(partial, 2, 32);
        if (j == 0)
            y[tok * D_INNER + h * PDIM + p] = partial + Dh * xv;
        __syncthreads();
    }
}

// ---------------------------------------------------------------- gate + RMSnorm
__global__ __launch_bounds__(256)
void gatenorm_kernel(const float* __restrict__ zx,   // z in cols [0,D_INNER)
                     const float* __restrict__ y,    // (TOKENS,D_INNER)
                     const float* __restrict__ nw,   // (D_INNER)
                     unsigned short* __restrict__ obf) { // bf16 (TOKENS,D_INNER)
    const int tok = blockIdx.x;
    __shared__ float red[256];
    __shared__ float g[D_INNER];
    float sum = 0.f;
    for (int c = threadIdx.x; c < D_INNER; c += 256) {
        float z  = zx[(size_t)tok * NPROJ + c];
        float gv = y[(size_t)tok * D_INNER + c] * (z / (1.f + __expf(-z)));
        g[c] = gv;
        sum += gv * gv;
    }
    red[threadIdx.x] = sum;
    __syncthreads();
    for (int s = 128; s > 0; s >>= 1) {
        if (threadIdx.x < s) red[threadIdx.x] += red[threadIdx.x + s];
        __syncthreads();
    }
    float inv = rsqrtf(red[0] / (float)D_INNER + 1e-5f);
    for (int c = threadIdx.x; c < D_INNER; c += 256)
        obf[(size_t)tok * D_INNER + c] = f32_to_bf16_rne(g[c] * inv * nw[c]);
}

// ---------------------------------------------------------------- mean over L
__global__ void mean_kernel(const float* __restrict__ x, float* __restrict__ xm) {
    int idx = blockIdx.x * blockDim.x + threadIdx.x;
    if (idx >= BATCH * D_MODEL) return;
    int d = idx % D_MODEL, b = idx / D_MODEL;
    float s = 0.f;
    for (int l = 0; l < SEQ; ++l)
        s += x[((size_t)(b * SEQ + l)) * D_MODEL + d];
    xm[idx] = s * (1.f / (float)SEQ);
}

// ---------------------------------------------------------------- output heads
#define OUT_EMO 7
#define OUT_SEN 3
__global__ __launch_bounds__(256)
void heads_kernel(const float* __restrict__ xm,
                  const float* __restrict__ we, const float* __restrict__ be,
                  const float* __restrict__ wsn, const float* __restrict__ bs,
                  float* __restrict__ out) {
    int o = blockIdx.x;  // 0..39 : 28 emo then 12 sen
    const float* W; const float* bias; int b, oi; float* dst;
    if (o < BATCH * OUT_EMO) {
        b = o / OUT_EMO; oi = o % OUT_EMO;
        W = we + (size_t)oi * D_MODEL; bias = be + oi;
        dst = out + b * OUT_EMO + oi;
    } else {
        int oo = o - BATCH * OUT_EMO;
        b = oo / OUT_SEN; oi = oo % OUT_SEN;
        W = wsn + (size_t)oi * D_MODEL; bias = bs + oi;
        dst = out + BATCH * OUT_EMO + b * OUT_SEN + oi;
    }
    __shared__ float red[256];
    float s = 0.f;
    for (int d = threadIdx.x; d < D_MODEL; d += 256)
        s += xm[(size_t)b * D_MODEL + d] * W[d];
    red[threadIdx.x] = s;
    __syncthreads();
    for (int st = 128; st > 0; st >>= 1) {
        if (threadIdx.x < st) red[threadIdx.x] += red[threadIdx.x + st];
        __syncthreads();
    }
    if (threadIdx.x == 0) *dst = red[0] + *bias;
}

// ================================================================ host
extern "C" void kernel_launch(void* const* d_in, const int* in_sizes, int n_in,
                              void* d_out, int out_size, void* d_ws, size_t ws_size,
                              hipStream_t stream) {
    (void)in_sizes; (void)n_in; (void)out_size; (void)ws_size;
    const float* x_in  = (const float*)d_in[0];
    const float* Wi    = (const float*)d_in[1];   // (5, NPROJ, D_MODEL)
    const float* cw    = (const float*)d_in[2];   // (5, CONV_DIM, 4)
    const float* cb    = (const float*)d_in[3];   // (5, CONV_DIM)
    const float* dtb   = (const float*)d_in[4];   // (5, HEADS)
    const float* alog  = (const float*)d_in[5];   // (5, HEADS)
    const float* Dp    = (const float*)d_in[6];   // (5, HEADS)
    const float* nw    = (const float*)d_in[7];   // (5, D_INNER)
    const float* Wo    = (const float*)d_in[8];   // (5, D_MODEL, D_INNER)
    const float* we    = (const float*)d_in[9];
    const float* be    = (const float*)d_in[10];
    const float* wsn   = (const float*)d_in[11];
    const float* bs    = (const float*)d_in[12];
    float* out = (float*)d_out;

    // workspace carve-up (256B aligned)
    char* ws = (char*)d_ws;
    size_t off = 0;
    auto carve = [&](size_t bytes) -> char* {
        char* p = ws + off;
        off = (off + bytes + 255) & ~(size_t)255;
        return p;
    };
    unsigned short* x_bf  = (unsigned short*)carve((size_t)TOKENS * D_MODEL * 2);
    float*          x_f32 = (float*)carve((size_t)TOKENS * D_MODEL * 4);
    unsigned short* Wi_bf = (unsigned short*)carve((size_t)N_LAYERS * NPROJ * D_MODEL * 2);
    unsigned short* Wo_bf = (unsigned short*)carve((size_t)N_LAYERS * D_MODEL * D_INNER * 2);
    float*          zx    = (float*)carve((size_t)TOKENS * NPROJ * 4);
    float*          xbc   = (float*)carve((size_t)TOKENS * CONV_DIM * 4);
    float*          dtv   = (float*)carve((size_t)TOKENS * HEADS * 4);
    float*          ybuf  = (float*)carve((size_t)TOKENS * D_INNER * 4);
    unsigned short* yn_bf = (unsigned short*)carve((size_t)TOKENS * D_INNER * 2);
    float*          xm    = (float*)carve((size_t)BATCH * D_MODEL * 4);

    // fp32 -> bf16 conversions (activations + all layer weights)
    {
        int n = TOKENS * D_MODEL;
        cvt_bf16_kernel<<<(n + 255) / 256, 256, 0, stream>>>(x_in, x_bf, n);
        n = N_LAYERS * NPROJ * D_MODEL;
        cvt_bf16_kernel<<<(n + 255) / 256, 256, 0, stream>>>(Wi, Wi_bf, n);
        n = N_LAYERS * D_MODEL * D_INNER;
        cvt_bf16_kernel<<<(n + 255) / 256, 256, 0, stream>>>(Wo, Wo_bf, n);
    }

    for (int L = 0; L < N_LAYERS; ++L) {
        const unsigned short* WiL = Wi_bf + (size_t)L * NPROJ * D_MODEL;
        const unsigned short* WoL = Wo_bf + (size_t)L * D_MODEL * D_INNER;

        // zxbcdt = x @ Wi^T  (M=4096, N=4256, K=1024)
        {
            int waves = (TOKENS / (16 * MT)) * (NPROJ / (16 * NT)); // 64*133
            int blocks = (waves + 7) / 8;
            wmma_gemm_bf16_kernel<<<blocks, 256, 0, stream>>>(
                x_bf, WiL, zx, nullptr, TOKENS, NPROJ, D_MODEL, 0);
        }
        // depthwise causal conv + SiLU
        {
            int n = TOKENS * CONV_DIM;
            conv_silu_kernel<<<(n + 255) / 256, 256, 0, stream>>>(
                zx, cw + (size_t)L * CONV_DIM * D_CONV, cb + (size_t)L * CONV_DIM, xbc);
        }
        // dt = softplus(raw + bias)
        {
            int n = TOKENS * HEADS;
            dt_kernel<<<(n + 255) / 256, 256, 0, stream>>>(
                zx, dtb + (size_t)L * HEADS, dtv);
        }
        // selective scan
        scan_kernel<<<BATCH * HEADS, 256, 0, stream>>>(
            xbc, dtv, alog + (size_t)L * HEADS, Dp + (size_t)L * HEADS, ybuf);
        // y * silu(z), RMS-norm, -> bf16
        gatenorm_kernel<<<TOKENS, 256, 0, stream>>>(
            zx, ybuf, nw + (size_t)L * D_INNER, yn_bf);
        // x = selu(yn @ Wo^T)  (M=4096, N=1024, K=2048); writes f32 + bf16
        {
            int waves = (TOKENS / (16 * MT)) * (D_MODEL / (16 * NT)); // 64*32
            int blocks = (waves + 7) / 8;
            wmma_gemm_bf16_kernel<<<blocks, 256, 0, stream>>>(
                yn_bf, WoL, x_f32, x_bf, TOKENS, D_MODEL, D_INNER, 1);
        }
    }

    // mean over sequence, then heads
    {
        int n = BATCH * D_MODEL;
        mean_kernel<<<(n + 255) / 256, 256, 0, stream>>>(x_f32, xm);
    }
    heads_kernel<<<BATCH * (OUT_EMO + OUT_SEN), 256, 0, stream>>>(
        xm, we, be, wsn, bs, out);
}
